// MoE_81209241633272
// MI455X (gfx1250) — matrix operand order repaired and checked by reading, not compiled, not run on hardware
//
#include <hip/hip_runtime.h>

// ---------------- problem constants ----------------
#define T_TOK  4096      // B*S tokens
#define D_DIM  2048      // model dim
#define I_DIM  1024      // expert inter dim (shared expert split into 2 slices of 1024)
#define N_EXP  16
#define TOPK   4

// ---------------- tiling ----------------
#define TM     32        // token rows per block
#define TN     64        // N columns per pass (8 waves: 2(M) x 4(N) of 16x16)
#define KT     32        // K per WMMA step (16x16x32 bf16)
#define H_LD   (I_DIM + 8)   // padded row stride of h tile (16B aligned rows, bank spread)
#define S_LD   48            // padded column stride in staging (mult of 16 -> 32B align)

typedef __attribute__((ext_vector_type(16))) __bf16 v16bf;
typedef __attribute__((ext_vector_type(8)))  __bf16 v8bf;
typedef __attribute__((ext_vector_type(2)))  __bf16 v2bf;
typedef __attribute__((ext_vector_type(8)))  float  v8f;
typedef __attribute__((ext_vector_type(4)))  float  v4f;

__device__ __forceinline__ v16bf cat16(v8bf lo, v8bf hi) {
    return __builtin_shufflevector(lo, hi, 0,1,2,3,4,5,6,7,8,9,10,11,12,13,14,15);
}

// B fragment: lane (n = lane&15) holds column n, K 0..15 (lane<16) or 16..31 (lane>=16),
// contiguous when staged as [n][k] column-major with stride S_LD.
__device__ __forceinline__ v16bf load_bfrag(const __bf16* p) {
    v8bf lo = *(const v8bf*)(p);
    v8bf hi = *(const v8bf*)(p + 8);
    return cat16(lo, hi);
}

// A fragment: lane<16 holds row, K 0..15; lane>=16 holds K 16..31.
// p already includes the +16 element offset for the upper half-lanes.
__device__ __forceinline__ v16bf load_afrag(const __bf16* p) {
    v8bf lo = *(const v8bf*)(p);
    v8bf hi = *(const v8bf*)(p + 8);
    return cat16(lo, hi);
}

// Stage a KT x 64 f32 weight tile (row-major, row stride ld) into LDS as bf16
// column-major [n][k].  Thread map: kp = tid>>4 (16 K-pairs), ng4 = (tid&15)*4
// (16 groups of 4 columns).  Coalesced b128 global reads, packed b32 LDS stores.
__device__ __forceinline__ void stage_w(__bf16* __restrict__ sbuf,
                                        const float* __restrict__ w, long long ld,
                                        int k, int ncol, int kp, int ng4) {
    const float* g0 = w + (size_t)(k + 2 * kp) * ld + ncol + ng4;
    const float* g1 = g0 + ld;
    v4f r0 = *(const v4f*)g0;
    v4f r1 = *(const v4f*)g1;
    #pragma unroll
    for (int j = 0; j < 4; ++j) {
        v2bf p;
        p[0] = (__bf16)r0[j];
        p[1] = (__bf16)r1[j];
        *(v2bf*)(sbuf + (size_t)(ng4 + j) * S_LD + 2 * kp) = p;
    }
}

// ---------------- gate: softmax + top-4 -> dense coef[T][16] ----------------
__global__ __launch_bounds__(128)
void gate_kernel(const float* __restrict__ x, const float* __restrict__ gw,
                 float* __restrict__ coef) {
    int tok = blockIdx.x;
    const float* xr = x + (size_t)tok * D_DIM;
    __shared__ float part[8][N_EXP];
    __shared__ float logits[N_EXP];
    int e = threadIdx.x & 15;
    int p = threadIdx.x >> 4;                   // 8 chunks of 256
    float s = 0.f;
    int j0 = p * 256;
    for (int j = j0; j < j0 + 256; ++j) s += xr[j] * gw[j * N_EXP + e];
    part[p][e] = s;
    __syncthreads();
    if (threadIdx.x < N_EXP) {
        float t = 0.f;
        for (int i = 0; i < 8; ++i) t += part[i][threadIdx.x];
        logits[threadIdx.x] = t;
    }
    __syncthreads();
    if (threadIdx.x == 0) {
        float mx = -1e30f;
        for (int i = 0; i < N_EXP; ++i) mx = fmaxf(mx, logits[i]);
        float pr[N_EXP], se = 0.f;
        for (int i = 0; i < N_EXP; ++i) { pr[i] = __expf(logits[i] - mx); se += pr[i]; }
        float inv = 1.f / se;
        for (int i = 0; i < N_EXP; ++i) pr[i] *= inv;
        float cf[N_EXP];
        for (int i = 0; i < N_EXP; ++i) cf[i] = 0.f;
        for (int k = 0; k < TOPK; ++k) {        // argmax picks first on ties (matches top_k)
            int bi = 0; float bv = -1.f;
            for (int i = 0; i < N_EXP; ++i) if (pr[i] > bv) { bv = pr[i]; bi = i; }
            cf[bi] = bv; pr[bi] = -2.f;
        }
        for (int i = 0; i < N_EXP; ++i) coef[(size_t)tok * N_EXP + i] = cf[i];
    }
}

// ---------------- deterministic per-expert token lists (one wave per expert) ----
__global__ __launch_bounds__(32)
void build_lists(const float* __restrict__ coef, int* __restrict__ list,
                 int* __restrict__ cnt) {
    int e = blockIdx.x;
    int lane = threadIdx.x;
    int c = 0;
    for (int t0 = 0; t0 < T_TOK; t0 += 32) {
        int t = t0 + lane;
        bool sel = coef[(size_t)t * N_EXP + e] > 0.f;
        unsigned m = (unsigned)__ballot(sel);
        int off = __popc(m & ((1u << lane) - 1u));
        if (sel) list[e * T_TOK + c + off] = t;
        c += __popc(m);
    }
    if (lane == 0) cnt[e] = c;
}

// ---------------- x -> bf16 (8 elements/thread, b128 in, b128 out) -------------
__global__ __launch_bounds__(256)
void cvt_kernel(const float* __restrict__ x, __bf16* __restrict__ xb, int n) {
    int i = (blockIdx.x * 256 + threadIdx.x) * 8;
    if (i < n) {
        v4f a = *(const v4f*)(x + i);
        v4f b = *(const v4f*)(x + i + 4);
        v8bf o;
        #pragma unroll
        for (int j = 0; j < 4; ++j) { o[j] = (__bf16)a[j]; o[4 + j] = (__bf16)b[j]; }
        *(v8bf*)(xb + i) = o;
    }
}

// ---------------- fused gated MLP: h = silu(x@Wa)*(x@Wb); out (+)= coef * h@Wc ----
// Wa/Wb: [D_DIM][*] row-major, row stride ldAB, per-expert stride strideEab elements.
// Wc:    [I_DIM][D_DIM] row-major, per-expert stride strideEc.
// list==nullptr -> identity token list over all T_TOK tokens, coef = 1.
// mode 0: plain store; mode 1: global atomic f32 add.
__global__ __launch_bounds__(256)
void moe_mlp_kernel(const __bf16* __restrict__ xb,
                    const float* __restrict__ Wa, const float* __restrict__ Wb,
                    long long ldAB, long long strideEab,
                    const float* __restrict__ Wc, long long strideEc,
                    const int* __restrict__ list, const int* __restrict__ cnt,
                    const float* __restrict__ coef,
                    float* __restrict__ out, int mode) {
    __shared__ __bf16 hTile[TM * H_LD];            // ~66 KB
    __shared__ __bf16 stage[2][2 * TN * S_LD];     // double-buffered, 24 KB
    __shared__ int   tokLds[TM];
    __shared__ float cfLds[TM];
    __shared__ char  okLds[TM];

    const int eIdx = blockIdx.y;
    const int rowsBase = blockIdx.x * TM;
    int nRows = T_TOK;
    const int* myList = nullptr;
    if (list) {
        nRows = cnt[eIdx];
        if (rowsBase >= nRows) return;            // uniform early exit
        myList = list + eIdx * T_TOK;
    }
    const float* wa = Wa + (size_t)eIdx * strideEab;
    const float* wb = Wb + (size_t)eIdx * strideEab;
    const float* wc = Wc + (size_t)eIdx * strideEc;

    const int tid = threadIdx.x;
    if (tid < TM) {
        int r = rowsBase + tid;
        bool ok; int tok;
        if (myList) { ok = (r < nRows); tok = ok ? myList[r] : 0; }
        else        { ok = true;        tok = r; }
        tokLds[tid] = tok;
        okLds[tid]  = ok ? 1 : 0;
        cfLds[tid]  = ok ? (myList ? coef[(size_t)tok * N_EXP + eIdx] : 1.0f) : 0.0f;
    }
    __syncthreads();

    const int lane = tid & 31;
    const int wave = tid >> 5;
    const int wm   = wave >> 2;       // 0..1  (M tile)
    const int wn   = wave & 3;        // 0..3  (N tile)
    const int nlo  = lane & 15;       // N index within 16 / A row within 16
    const int hi   = lane >> 4;       // K half selector

    const int myRow = wm * 16 + nlo;
    const size_t aBase = (size_t)tokLds[myRow] * D_DIM;

    // staging thread map: 16 K-pairs x 16 column groups of 4
    const int kp  = tid >> 4;          // 0..15
    const int ng4 = (tid & 15) * 4;    // 0..60
    const int boff = wn * 16 + nlo;    // B-fragment column within the 64-wide tile

    // ---------------- Phase A: h = silu(x@Wa) * (x@Wb) ----------------
    {
        const int NK = D_DIM / KT;                  // 64 K-steps per column block
        const int S  = (I_DIM / TN) * NK;           // 1024 pipeline steps
        v8f acc1 = (v8f)0.0f, acc3 = (v8f)0.0f;
        // pipeline prologue: stage step 0 into buffer 0
        stage_w(stage[0],             wa, ldAB, 0, 0, kp, ng4);
        stage_w(stage[0] + TN * S_LD, wb, ldAB, 0, 0, kp, ng4);
        for (int s = 0; s < S; ++s) {
            const int nc = (s / NK) * TN;
            const int k  = (s % NK) * KT;
            __bf16* cur = stage[s & 1];
            __syncthreads();                        // cur writes visible; prev reads done
            if (s + 1 < S) {                        // stage next step into other buffer
                int nc1 = ((s + 1) / NK) * TN;
                int k1  = ((s + 1) % NK) * KT;
                __bf16* nxt = stage[(s + 1) & 1];
                stage_w(nxt,             wa, ldAB, k1, nc1, kp, ng4);
                stage_w(nxt + TN * S_LD, wb, ldAB, k1, nc1, kp, ng4);
            }
            v16bf afrag = load_afrag(xb + aBase + k + hi * 16);
            v16bf b1f   = load_bfrag(cur + boff * S_LD + hi * 16);
            v16bf b3f   = load_bfrag(cur + TN * S_LD + boff * S_LD + hi * 16);
            acc1 = __builtin_amdgcn_wmma_f32_16x16x32_bf16(false, afrag, false, b1f,
                                                           (short)0, acc1, false, false);
            acc3 = __builtin_amdgcn_wmma_f32_16x16x32_bf16(false, afrag, false, b3f,
                                                           (short)0, acc3, false, false);
            if ((s % NK) == NK - 1) {               // end of K sweep: silu*mul -> hTile
                #pragma unroll
                for (int i = 0; i < 8; ++i) {       // C layout: VGPR i -> row 16*wm+i+8*hi
                    int r = wm * 16 + i + hi * 8;
                    int c = nc + wn * 16 + nlo;
                    float a = acc1[i];
                    float hval = a * __builtin_amdgcn_rcpf(1.f + __expf(-a)) * acc3[i];
                    hTile[r * H_LD + c] = (__bf16)hval;
                }
                acc1 = (v8f)0.0f;
                acc3 = (v8f)0.0f;
            }
        }
    }
    __syncthreads();   // hTile complete; last stage reads done

    // ---------------- Phase B: out (+)= coef * (h @ Wc) ----------------
    {
        const int NK = I_DIM / KT;                  // 32
        const int S  = (D_DIM / TN) * NK;           // 1024
        v8f acc = (v8f)0.0f;
        stage_w(stage[0], wc, D_DIM, 0, 0, kp, ng4);
        for (int s = 0; s < S; ++s) {
            const int nc = (s / NK) * TN;
            const int k  = (s % NK) * KT;
            __bf16* cur = stage[s & 1];
            __syncthreads();
            if (s + 1 < S) {
                int nc1 = ((s + 1) / NK) * TN;
                int k1  = ((s + 1) % NK) * KT;
                stage_w(stage[(s + 1) & 1], wc, D_DIM, k1, nc1, kp, ng4);
            }
            v16bf afrag = load_afrag(hTile + myRow * H_LD + k + hi * 16);
            v16bf bfrag = load_bfrag(cur + boff * S_LD + hi * 16);
            acc = __builtin_amdgcn_wmma_f32_16x16x32_bf16(false, afrag, false, bfrag,
                                                          (short)0, acc, false, false);
            if ((s % NK) == NK - 1) {
                #pragma unroll
                for (int i = 0; i < 8; ++i) {
                    int r = wm * 16 + i + hi * 8;
                    int c = nc + wn * 16 + nlo;
                    if (okLds[r]) {
                        float v = acc[i] * cfLds[r];
                        float* dst = out + (size_t)tokLds[r] * D_DIM + c;
                        if (mode) atomicAdd(dst, v);
                        else      *dst = v;
                    }
                }
                acc = (v8f)0.0f;
            }
        }
    }
}

// ---------------- host launcher ----------------
extern "C" void kernel_launch(void* const* d_in, const int* in_sizes, int n_in,
                              void* d_out, int out_size, void* d_ws, size_t ws_size,
                              hipStream_t stream) {
    const float* x      = (const float*)d_in[0];   // [4,1024,2048]
    const float* gate_w = (const float*)d_in[1];   // [2048,16]
    const float* w1     = (const float*)d_in[2];   // [16,2048,1024]
    const float* w2     = (const float*)d_in[3];   // [16,1024,2048]
    const float* w3     = (const float*)d_in[4];   // [16,2048,1024]
    const float* ws1    = (const float*)d_in[5];   // [2048,2048]
    const float* ws2    = (const float*)d_in[6];   // [2048,2048]
    const float* ws3    = (const float*)d_in[7];   // [2048,2048]
    float* out = (float*)d_out;

    // workspace layout
    const size_t XB_BYTES   = (size_t)T_TOK * D_DIM * 2;        // 16 MB bf16 x
    const size_t COEF_BYTES = (size_t)T_TOK * N_EXP * 4;        // dense gate coefs
    char* ws = (char*)d_ws;
    __bf16* xb  = (__bf16*)(ws);
    float*  coef = (float*)(ws + XB_BYTES);
    int*    cnt  = (int*)(ws + XB_BYTES + COEF_BYTES);
    int*    list = (int*)(ws + XB_BYTES + COEF_BYTES + 64);

    // 1. gating (softmax + top-4)
    gate_kernel<<<dim3(T_TOK), dim3(128), 0, stream>>>(x, gate_w, coef);
    // 2. deterministic per-expert token lists
    build_lists<<<dim3(N_EXP), dim3(32), 0, stream>>>(coef, list, cnt);
    // 3. x -> bf16
    cvt_kernel<<<dim3((T_TOK * D_DIM) / (256 * 8)), dim3(256), 0, stream>>>(
        x, xb, T_TOK * D_DIM);
    // 4. shared expert, columns [0,1024): plain store initializes d_out
    moe_mlp_kernel<<<dim3(T_TOK / TM, 1), dim3(256), 0, stream>>>(
        xb, ws1, ws3, (long long)(2 * I_DIM), 0,
        ws2, 0, nullptr, nullptr, nullptr, out, 0);
    // 5. shared expert, columns [1024,2048): accumulate
    moe_mlp_kernel<<<dim3(T_TOK / TM, 1), dim3(256), 0, stream>>>(
        xb, ws1 + I_DIM, ws3 + I_DIM, (long long)(2 * I_DIM), 0,
        ws2 + (size_t)I_DIM * D_DIM, 0, nullptr, nullptr, nullptr, out, 1);
    // 6. routed experts: gather rows per expert, scatter-accumulate with gate coef
    moe_mlp_kernel<<<dim3(T_TOK / TM, N_EXP), dim3(256), 0, stream>>>(
        xb, w1, w3, (long long)I_DIM, (long long)D_DIM * I_DIM,
        w2, (long long)I_DIM * D_DIM, list, cnt, coef, out, 1);
}
